// GNN_5334349382373
// MI455X (gfx1250) — matrix operand order repaired and checked by reading, not compiled, not run on hardware
//
#include <hip/hip_runtime.h>
#include <hip/hip_bf16.h>

#define N_NODES 40000      // divisible by 64: GEMM tiling is exact, no tail path
#define N_EDGES 640000
#define DIM     128
#define NGRAPH  64
#define NCLASS  10
#define LN_EPS  1e-5f

typedef __attribute__((ext_vector_type(16))) _Float16 v16h;
typedef __attribute__((ext_vector_type(8)))  float    v8f;

// ---------------------------------------------------------------- utilities
__global__ void fill_kernel(float* __restrict__ p, float v, int n) {
    int i = blockIdx.x * blockDim.x + threadIdx.x;
    if (i < n) p[i] = v;
}

// deg[col[e]] += 1  (deg pre-filled with 1.0 for the self-loop)
__global__ void deg_kernel(const int* __restrict__ col, float* __restrict__ deg) {
    int e = blockIdx.x * blockDim.x + threadIdx.x;
    if (e < N_EDGES) atomicAdd(&deg[col[e]], 1.0f);
}

// in-place deg -> dinv = rsqrt(max(deg,1))
__global__ void dinv_kernel(float* __restrict__ deg) {
    int i = blockIdx.x * blockDim.x + threadIdx.x;
    if (i < N_NODES) deg[i] = rsqrtf(fmaxf(deg[i], 1.0f));
}

// ------------------------------------------------- GEMM: out[N,128] = A[N,128] @ W[128,128]
// Block: 256 threads = 8 waves; covers exactly 64 rows x 128 cols (N % 64 == 0,
// so no bounds checks anywhere). Wave w owns column tile w, loops 4 row tiles.
// W stored TRANSPOSED in LDS so each lane's B fragment is 16 contiguous halfs
// -> hot loop is 4x ds_load_b128 per v_wmma_f32_16x16x32_f16.
__global__ __launch_bounds__(256) void gemm_wmma_kernel(
    const float* __restrict__ A, const float* __restrict__ W,
    float* __restrict__ out) {
    __shared__ _Float16 sWt[DIM * DIM];  // 32 KB, sWt[c*DIM + k] = W[k*DIM + c]
    __shared__ _Float16 sA[64 * DIM];    // 16 KB, row-major

    const int tid     = threadIdx.x;
    const int rowBase = blockIdx.x * 64;

    for (int i = tid; i < DIM * DIM; i += 256) {
        int k = i >> 7, c = i & 127;
        sWt[c * DIM + k] = (_Float16)W[i];
    }
    for (int i = tid; i < 64 * DIM; i += 256) {
        int r = rowBase + (i >> 7);
        sA[i] = (_Float16)A[r * DIM + (i & 127)];
    }
    __syncthreads();

    const int lane  = tid & 31;
    const int wave  = tid >> 5;
    const int group = lane >> 4;   // 0 or 1 (half-wave)
    const int lr    = lane & 15;
    const int ncol  = wave * 16;   // column-tile base

    for (int mt = 0; mt < 4; ++mt) {
        v8f c = {};
        const int arow = mt * 16 + lr;  // lane holds A row lane%16 of this tile
        for (int kk = 0; kk < DIM; kk += 32) {
            v16h a, b;
#pragma unroll
            for (int i = 0; i < 16; ++i) {
                // A 16x32 per-lane K layout (contiguous, matches B):
                // lanes 0-15: K in {0..15}; lanes 16-31: K in {16..31}
                int ka = kk + group * 16 + i;
                a[i] = sA[arow * DIM + ka];
                // B 32x16: lane holds column lr; K = kk + 16*group + i (contiguous)
                b[i] = sWt[(ncol + lr) * DIM + kk + group * 16 + i];
            }
            c = __builtin_amdgcn_wmma_f32_16x16x32_f16(
                    /*neg_a=*/false, a, /*neg_b=*/false, b,
                    /*c_mod=*/(short)0, c, /*reuse_a=*/false, /*reuse_b=*/false);
        }
        // D: VGPR i -> row i + 8*group, col lr; straight-line coalesced stores
#pragma unroll
        for (int i = 0; i < 8; ++i) {
            int r = rowBase + mt * 16 + i + group * 8;
            out[r * DIM + ncol + lr] = c[i];
        }
    }
}

// -------------------------------- edge scatter: agg[col] += t[row] * dinv[row]*dinv[col]
// One wave per edge; lane handles 4 contiguous features (float4 gather, 4 atomic f32 adds).
__global__ __launch_bounds__(256) void agg_kernel(
    const int* __restrict__ row, const int* __restrict__ col,
    const float* __restrict__ dinv, const float* __restrict__ t,
    float* __restrict__ agg) {
    int e    = (blockIdx.x * blockDim.x + threadIdx.x) >> 5;
    int lane = threadIdx.x & 31;
    if (e >= N_EDGES) return;
    int r = row[e], cN = col[e];
    float nrm = dinv[r] * dinv[cN];
    float4 v = ((const float4*)(t + (size_t)r * DIM))[lane];
    float* dst = agg + (size_t)cN * DIM + lane * 4;
    atomicAdd(dst + 0, v.x * nrm);
    atomicAdd(dst + 1, v.y * nrm);
    atomicAdd(dst + 2, v.z * nrm);
    atomicAdd(dst + 3, v.w * nrm);
}

// -------- fused: v = agg + t*dinv^2 (self-loop) + bias; out = relu(LN(v)*gamma+beta)
// One wave per node; width-32 shuffle reductions (wave32).
__global__ __launch_bounds__(256) void ln_relu_kernel(
    const float* __restrict__ agg, const float* __restrict__ t,
    const float* __restrict__ dinv, const float* __restrict__ bias,
    const float* __restrict__ gamma, const float* __restrict__ beta,
    float* __restrict__ out) {
    int node = (blockIdx.x * blockDim.x + threadIdx.x) >> 5;
    int lane = threadIdx.x & 31;
    if (node >= N_NODES) return;
    float di   = dinv[node];
    float self = di * di;
    float4 a  = ((const float4*)(agg + (size_t)node * DIM))[lane];
    float4 tv = ((const float4*)(t   + (size_t)node * DIM))[lane];
    float4 bv = ((const float4*)bias)[lane];
    float v0 = a.x + tv.x * self + bv.x;
    float v1 = a.y + tv.y * self + bv.y;
    float v2 = a.z + tv.z * self + bv.z;
    float v3 = a.w + tv.w * self + bv.w;
    float s  = v0 + v1 + v2 + v3;
    float ss = v0 * v0 + v1 * v1 + v2 * v2 + v3 * v3;
    for (int off = 16; off >= 1; off >>= 1) {
        s  += __shfl_xor(s,  off, 32);
        ss += __shfl_xor(ss, off, 32);
    }
    float mu  = s * (1.0f / DIM);
    float var = ss * (1.0f / DIM) - mu * mu;
    float inv = rsqrtf(var + LN_EPS);
    float4 g  = ((const float4*)gamma)[lane];
    float4 be = ((const float4*)beta)[lane];
    float4 o;
    o.x = fmaxf((v0 - mu) * inv * g.x + be.x, 0.0f);
    o.y = fmaxf((v1 - mu) * inv * g.y + be.y, 0.0f);
    o.z = fmaxf((v2 - mu) * inv * g.z + be.z, 0.0f);
    o.w = fmaxf((v3 - mu) * inv * g.w + be.w, 0.0f);
    ((float4*)(out + (size_t)node * DIM))[lane] = o;
}

// ---------------------------------------- global mean-pool accumulation (atomic)
__global__ __launch_bounds__(256) void pool_kernel(
    const float* __restrict__ h, const int* __restrict__ batch,
    float* __restrict__ sums, float* __restrict__ cnts) {
    int node = (blockIdx.x * blockDim.x + threadIdx.x) >> 5;
    int lane = threadIdx.x & 31;
    if (node >= N_NODES) return;
    int g = batch[node];
    float4 v = ((const float4*)(h + (size_t)node * DIM))[lane];
    float* dst = sums + (size_t)g * DIM + lane * 4;
    atomicAdd(dst + 0, v.x);
    atomicAdd(dst + 1, v.y);
    atomicAdd(dst + 2, v.z);
    atomicAdd(dst + 3, v.w);
    if (lane == 0) atomicAdd(&cnts[g], 1.0f);
}

// out[g,c] = (sums[g]·Wl[:,c]) / max(cnt,1) + bl[c]
__global__ void final_kernel(const float* __restrict__ sums, const float* __restrict__ cnts,
                             const float* __restrict__ Wl, const float* __restrict__ bl,
                             float* __restrict__ out) {
    int idx = blockIdx.x * blockDim.x + threadIdx.x;
    if (idx >= NGRAPH * NCLASS) return;
    int g = idx / NCLASS, c = idx % NCLASS;
    float acc = 0.0f;
    for (int k = 0; k < DIM; ++k) acc += sums[g * DIM + k] * Wl[k * NCLASS + c];
    out[idx] = acc / fmaxf(cnts[g], 1.0f) + bl[c];
}

// ================================================================== launch
extern "C" void kernel_launch(void* const* d_in, const int* in_sizes, int n_in,
                              void* d_out, int out_size, void* d_ws, size_t ws_size,
                              hipStream_t stream) {
    const float* x     = (const float*)d_in[0];
    const int*   eidx  = (const int*)d_in[1];      // [2,E] flat
    const int*   batch = (const int*)d_in[2];
    const float* W1    = (const float*)d_in[3];
    const float* b1    = (const float*)d_in[4];
    const float* W2    = (const float*)d_in[5];
    const float* b2    = (const float*)d_in[6];
    const float* g1    = (const float*)d_in[7];
    const float* be1   = (const float*)d_in[8];
    const float* g2    = (const float*)d_in[9];
    const float* be2   = (const float*)d_in[10];
    const float* Wl    = (const float*)d_in[11];
    const float* bl    = (const float*)d_in[12];
    float*       out   = (float*)d_out;

    const int* row = eidx;
    const int* col = eidx + N_EDGES;

    // workspace layout (256B aligned)
    char*  wsb = (char*)d_ws;
    size_t off = 0;
    auto alloc = [&](size_t bytes) -> float* {
        float* p = (float*)(wsb + off);
        off = (off + bytes + 255) & ~(size_t)255;
        return p;
    };
    float* dinv = alloc((size_t)N_NODES * 4);            // deg, then in-place dinv
    float* bufA = alloc((size_t)N_NODES * DIM * 4);      // GEMM outputs
    float* bufB = alloc((size_t)N_NODES * DIM * 4);      // aggregation / activations
    float* sums = alloc((size_t)NGRAPH * DIM * 4);
    float* cnts = alloc((size_t)NGRAPH * 4);
    (void)ws_size; (void)in_sizes; (void)n_in; (void)out_size;

    const int TB = 256;
    const int nFeat      = N_NODES * DIM;
    const int gFill      = (nFeat + TB - 1) / TB;
    const int gEdgeThr   = (N_EDGES + TB - 1) / TB;
    const int gNode      = (N_NODES + TB - 1) / TB;
    const int gGemm      = N_NODES / 64;                   // exact (625 blocks)
    const int gEdgeWave  = (N_EDGES * 32 + TB - 1) / TB;   // wave per edge
    const int gNodeWave  = (N_NODES * 32 + TB - 1) / TB;   // wave per node

    // degree + symmetric normalization
    fill_kernel<<<gNode, TB, 0, stream>>>(dinv, 1.0f, N_NODES);       // self-loop
    deg_kernel<<<gEdgeThr, TB, 0, stream>>>(col, dinv);
    dinv_kernel<<<gNode, TB, 0, stream>>>(dinv);

    // ---- layer 1
    gemm_wmma_kernel<<<gGemm, TB, 0, stream>>>(x, W1, bufA);
    fill_kernel<<<gFill, TB, 0, stream>>>(bufB, 0.0f, nFeat);
    agg_kernel<<<gEdgeWave, TB, 0, stream>>>(row, col, dinv, bufA, bufB);
    ln_relu_kernel<<<gNodeWave, TB, 0, stream>>>(bufB, bufA, dinv, b1, g1, be1, bufB);

    // ---- layer 2
    gemm_wmma_kernel<<<gGemm, TB, 0, stream>>>(bufB, W2, bufA);
    fill_kernel<<<gFill, TB, 0, stream>>>(bufB, 0.0f, nFeat);
    agg_kernel<<<gEdgeWave, TB, 0, stream>>>(row, col, dinv, bufA, bufB);
    ln_relu_kernel<<<gNodeWave, TB, 0, stream>>>(bufB, bufA, dinv, b2, g2, be2, bufB);

    // ---- pool + classifier head
    fill_kernel<<<(NGRAPH * DIM + TB - 1) / TB, TB, 0, stream>>>(sums, 0.0f, NGRAPH * DIM);
    fill_kernel<<<1, TB, 0, stream>>>(cnts, 0.0f, NGRAPH);
    pool_kernel<<<gNodeWave, TB, 0, stream>>>(bufB, batch, sums, cnts);
    final_kernel<<<(NGRAPH * NCLASS + TB - 1) / TB, TB, 0, stream>>>(sums, cnts, Wl, bl, out);
}
